// RNNLayer_3650722201927
// MI455X (gfx1250) — matrix operand order repaired and checked by reading, not compiled, not run on hardware
//
#include <hip/hip_runtime.h>

#define SEQ     8192
#define IN_DIM  1024
#define HID_DIM 1024
#define OUT_DIM 1024
#define WIDTH   2048
#define DEPTH   3
#define NWG     16      // workgroups in the persistent sequential kernel

typedef __attribute__((ext_vector_type(16))) __bf16 v16bf;
typedef __attribute__((ext_vector_type(8)))  __bf16 v8bf;
typedef __attribute__((ext_vector_type(8)))  float  v8f;

// ---------------------------------------------------------------- helpers
static __device__ __forceinline__ unsigned short f32_to_bf16(float f) {
  unsigned u = __float_as_uint(f);
  u += 0x7FFFu + ((u >> 16) & 1u);          // round-to-nearest-even
  return (unsigned short)(u >> 16);
}

static __device__ __forceinline__ float bf16_to_f32(unsigned short h) {
  return __uint_as_float(((unsigned)h) << 16);
}

// Sense-reversing grid barrier for the NWG persistent blocks.
static __device__ __forceinline__ void grid_barrier(unsigned* cnt, unsigned* gen,
                                                    unsigned* epoch) {
  __threadfence();              // release: make this thread's global stores visible
  __syncthreads();
  if (threadIdx.x == 0) {
    const unsigned e = *epoch;
    const unsigned prev =
        __hip_atomic_fetch_add(cnt, 1u, __ATOMIC_ACQ_REL, __HIP_MEMORY_SCOPE_AGENT);
    if (prev == (unsigned)(NWG - 1)) {
      __hip_atomic_store(cnt, 0u, __ATOMIC_RELAXED, __HIP_MEMORY_SCOPE_AGENT);
      __hip_atomic_store(gen, e + 1u, __ATOMIC_RELEASE, __HIP_MEMORY_SCOPE_AGENT);
    } else {
      while (__hip_atomic_load(gen, __ATOMIC_ACQUIRE, __HIP_MEMORY_SCOPE_AGENT) == e) {
        __builtin_amdgcn_s_sleep(1);
      }
    }
    *epoch = e + 1u;
  }
  __syncthreads();
  __threadfence();              // acquire: invalidate stale near caches
}

// ---------------------------------------------------------------- fp32 -> bf16 cast
__global__ void __launch_bounds__(256) cast_f32_bf16(const float* __restrict__ s,
                                                     unsigned short* __restrict__ d,
                                                     long long n) {
  long long i = (long long)blockIdx.x * blockDim.x + threadIdx.x;
  const long long stride = (long long)gridDim.x * blockDim.x;
  for (; i < n; i += stride) d[i] = f32_to_bf16(s[i]);
}

// ---------------------------------------------------------------- pack W into WMMA B-operand tiles
// dst element index = ((kt*ntN + nt)*32 + lane)*16 + j
// B 32x16 bf16 layout: lane L holds N = nt*16 + (L&15); K = kt*32 + ((L&16)?16:0) + j
__global__ void __launch_bounds__(256) pack_wmma_b(const float* __restrict__ W,
                                                   int row0, int col0, int ld,
                                                   int K, int N,
                                                   unsigned short* __restrict__ dst) {
  const int ntN = N >> 4;
  const long long total = (long long)K * N;
  long long i = (long long)blockIdx.x * blockDim.x + threadIdx.x;
  const long long stride = (long long)gridDim.x * blockDim.x;
  for (; i < total; i += stride) {
    const int j    = (int)(i & 15);
    const int lane = (int)((i >> 4) & 31);
    const long long tile = i >> 9;
    const int nt = (int)(tile % ntN);
    const int kt = (int)(tile / ntN);
    const int k  = kt * 32 + ((lane & 16) ? 16 : 0) + j;
    const int n  = nt * 16 + (lane & 15);
    dst[i] = f32_to_bf16(W[(long long)(row0 + k) * ld + (col0 + n)]);
  }
}

// ---------------------------------------------------------------- batched WMMA GEMM
// Out[M,N] = A(bf16,row-major,lda) @ Bpacked + bias(row broadcast), fp32 out.
// One wave per 16x16 output tile; grid = (M/16)*(N/16)/8 blocks of 8 waves.
__global__ void __launch_bounds__(256) wmma_gemm_bf16(const unsigned short* __restrict__ A,
                                                      int lda,
                                                      const unsigned short* __restrict__ Bp,
                                                      const float* __restrict__ bias,
                                                      float* __restrict__ Out, int ldo,
                                                      int ntN, int Ktiles) {
  const int lane = threadIdx.x & 31;
  const int gw   = blockIdx.x * 8 + (threadIdx.x >> 5);
  const int mt   = gw / ntN;
  const int nt   = gw % ntN;
  const int off  = (lane & 16) ? 16 : 0;    // A-operand K-offset per lane half
  const __bf16* Ab = reinterpret_cast<const __bf16*>(A);
  const long long arow = (long long)(mt * 16 + (lane & 15)) * lda;

  v8f acc = {0.f, 0.f, 0.f, 0.f, 0.f, 0.f, 0.f, 0.f};
  for (int kt = 0; kt < Ktiles; ++kt) {
    v16bf a, b;
    a.lo = *reinterpret_cast<const v8bf*>(Ab + arow + kt * 32 + off);
    a.hi = *reinterpret_cast<const v8bf*>(Ab + arow + kt * 32 + 8 + off);
    b = *reinterpret_cast<const v16bf*>(Bp + (((long long)kt * ntN + nt) * 32 + lane) * 16);
    acc = __builtin_amdgcn_wmma_f32_16x16x32_bf16(false, a, false, b, (short)0, acc,
                                                  false, false);
  }
  const int n = nt * 16 + (lane & 15);
  const float bs = bias ? bias[n] : 0.f;
  const int mbase = mt * 16 + ((lane & 16) ? 8 : 0);
#pragma unroll
  for (int r = 0; r < 8; ++r)
    Out[(long long)(mbase + r) * ldo + n] = acc[r] + bs;
}

// ---------------------------------------------------------------- d = bf[:1024] @ W0[1024:,:]
__global__ void __launch_bounds__(256) bias_w0h(const float* __restrict__ W0,
                                                const float* __restrict__ bf,
                                                float* __restrict__ d) {
  const int n = blockIdx.x * 256 + threadIdx.x;    // 0..2047
  if (n >= WIDTH) return;
  float acc = 0.f;
  for (int k = 0; k < HID_DIM; ++k)
    acc += bf[k] * W0[(long long)(IN_DIM + k) * WIDTH + n];
  d[n] = acc;
}

// C[t][n] += d[n] for t >= 1 (folds the fused-layer bias into the precomputed row)
__global__ void __launch_bounds__(256) add_d(float* __restrict__ C,
                                             const float* __restrict__ d) {
  const long long total = (long long)SEQ * WIDTH;
  const long long stride = (long long)gridDim.x * 256;
  for (long long i = (long long)blockIdx.x * 256 + threadIdx.x + WIDTH; i < total;
       i += stride)
    C[i] += d[i & (WIDTH - 1)];
}

// hidden_final = R[SEQ-1] @ Wf[:, :1024] + bf[:1024]   (fp32 weights, tiny GEMV)
__global__ void __launch_bounds__(256) final_hidden(const unsigned short* __restrict__ r,
                                                    const float* __restrict__ Wf,
                                                    const float* __restrict__ bf,
                                                    float* __restrict__ outh) {
  const int n = blockIdx.x * 256 + threadIdx.x;    // 0..1023
  if (n >= HID_DIM) return;
  float acc = bf[n];
  for (int k = 0; k < WIDTH; ++k)
    acc += bf16_to_f32(r[k]) * Wf[(long long)k * (HID_DIM + OUT_DIM) + n];
  outh[n] = acc;
}

// ---------------------------------------------------------------- sequential recurrence
// 16 persistent blocks x 8 waves = 128 waves = 128 n-tiles (N=2048).
// 4 uniform dependent layers per step (Wfused folds Wf_hidden @ W0h):
//   h0 = r_{t-1} @ Wfused + C'[t];  h1..h3 via Wmid;  R[t] = relu(h3).
// Activation broadcast into LDS uses the CDNA5 async-LDS path (ASYNCcnt).
__global__ void __launch_bounds__(256) rnn_sequential(const float* __restrict__ C,    // [SEQ][WIDTH] = X@W0x+b0 (+d for t>0)
                                                      const unsigned short* __restrict__ Wfus,
                                                      const unsigned short* __restrict__ Wm,
                                                      const float* __restrict__ bmid, // [DEPTH][WIDTH]
                                                      const unsigned short* __restrict__ Z, // 2048 bf16 zeros
                                                      unsigned short* __restrict__ S0,
                                                      unsigned short* __restrict__ S1,
                                                      unsigned short* __restrict__ R, // [SEQ][WIDTH] relu chain
                                                      unsigned* __restrict__ bar_cnt,
                                                      unsigned* __restrict__ bar_gen) {
  __shared__ __bf16 hbuf[WIDTH];
  const int tid  = threadIdx.x;
  const int lane = tid & 31;
  const int nt   = blockIdx.x * 8 + (tid >> 5);   // n-tile owned by this wave (0..127)
  const int off  = (lane & 16) ? 16 : 0;
  unsigned epoch = 0;

  const __bf16* hb = hbuf;
  const unsigned hbase = (unsigned)(unsigned long long)(const void*)hbuf; // LDS byte addr

  for (int t = 0; t < SEQ; ++t) {
#pragma unroll 1
    for (int l = 0; l < 4; ++l) {
      const unsigned short* src;
      const unsigned short* Bp;
      const float* bias;
      unsigned short* dst;
      if (l == 0) {
        src = (t == 0) ? Z : R + (long long)(t - 1) * WIDTH;
        Bp = Wfus; bias = C + (long long)t * WIDTH; dst = S0;
      } else if (l == 1) {
        src = S0; Bp = Wm; bias = bmid; dst = S1;
      } else if (l == 2) {
        src = S1; Bp = Wm + (size_t)WIDTH * WIDTH; bias = bmid + WIDTH; dst = S0;
      } else {
        src = S0; Bp = Wm + 2 * (size_t)WIDTH * WIDTH; bias = bmid + 2 * WIDTH;
        dst = R + (long long)t * WIDTH;
      }

      // Async broadcast of the 4KB activation vector into this WGP's LDS:
      // 256 lanes x async b128 (ASYNCcnt), then wait + workgroup barrier.
      {
        const unsigned lds = hbase + (unsigned)tid * 16u;
        const unsigned long long ga =
            (unsigned long long)src + (unsigned long long)tid * 16u;
        asm volatile("global_load_async_to_lds_b128 %0, %1, off"
                     :: "v"(lds), "v"(ga) : "memory");
        asm volatile("s_wait_asynccnt 0x0" ::: "memory");
      }
      __syncthreads();

      v8f acc = {0.f, 0.f, 0.f, 0.f, 0.f, 0.f, 0.f, 0.f};
      for (int kt = 0; kt < WIDTH / 32; ++kt) {
        v16bf a, b;
        a.lo = *reinterpret_cast<const v8bf*>(hb + kt * 32 + off);
        a.hi = *reinterpret_cast<const v8bf*>(hb + kt * 32 + 8 + off);
        b = *reinterpret_cast<const v16bf*>(Bp + (((long long)kt * 128 + nt) * 32 + lane) * 16);
        acc = __builtin_amdgcn_wmma_f32_16x16x32_bf16(false, a, false, b, (short)0, acc,
                                                      false, false);
      }
      if (lane < 16) {                 // D row 0 = result (all A rows identical)
        const int n = nt * 16 + lane;
        float v = fmaxf(acc[0] + bias[n], 0.0f);   // every layer output is consumed ReLU'd
        dst[n] = f32_to_bf16(v);
      }
      grid_barrier(bar_cnt, bar_gen, &epoch);
    }
  }
}

// ---------------------------------------------------------------- launch
extern "C" void kernel_launch(void* const* d_in, const int* in_sizes, int n_in,
                              void* d_out, int out_size, void* d_ws, size_t ws_size,
                              hipStream_t stream) {
  (void)in_sizes; (void)n_in; (void)out_size; (void)ws_size;

  const float* x    = (const float*)d_in[0];
  const float* W0   = (const float*)d_in[1];
  const float* b0   = (const float*)d_in[2];
  const float* Wmid = (const float*)d_in[3];
  const float* bmid = (const float*)d_in[4];
  const float* Wf   = (const float*)d_in[5];
  const float* bf   = (const float*)d_in[6];
  float* out = (float*)d_out;

  char* ws = (char*)d_ws;
  size_t off = 0;
  auto take = [&](size_t bytes) -> void* {
    void* p = ws + off;
    off += (bytes + 255) & ~(size_t)255;
    return p;
  };

  unsigned short* Xb    = (unsigned short*)take((size_t)SEQ * IN_DIM * 2);        // 16 MB
  float*          Cb    = (float*)take((size_t)SEQ * WIDTH * 4);                  // 64 MB
  unsigned short* Rb    = (unsigned short*)take((size_t)SEQ * WIDTH * 2);         // 32 MB
  unsigned short* pW0x  = (unsigned short*)take((size_t)IN_DIM * WIDTH * 2);      //  4 MB
  unsigned short* pW0h  = (unsigned short*)take((size_t)HID_DIM * WIDTH * 2);     //  4 MB
  unsigned short* pWm   = (unsigned short*)take((size_t)DEPTH * WIDTH * WIDTH * 2); // 24 MB
  unsigned short* pWfo  = (unsigned short*)take((size_t)WIDTH * OUT_DIM * 2);     //  4 MB
  unsigned short* Wfb   = (unsigned short*)take((size_t)WIDTH * (HID_DIM + OUT_DIM) * 2); // 8 MB
  float*          Wfus  = (float*)take((size_t)WIDTH * WIDTH * 4);                // 16 MB
  unsigned short* pWfus = (unsigned short*)take((size_t)WIDTH * WIDTH * 2);       //  8 MB
  float*          dvec  = (float*)take((size_t)WIDTH * 4);
  char* ctrl = (char*)take(256 + 3 * (size_t)WIDTH * 2);   // barrier + Z + S0 + S1
  unsigned* bar_cnt = (unsigned*)ctrl;
  unsigned* bar_gen = bar_cnt + 1;
  unsigned short* Zb = (unsigned short*)(ctrl + 256);
  unsigned short* S0 = Zb + WIDTH;
  unsigned short* S1 = S0 + WIDTH;

  // Zero barrier state + zero-vector (initial r_{-1}). Capture-safe.
  hipMemsetAsync(ctrl, 0, 256 + 3 * (size_t)WIDTH * 2, stream);

  // Cast activations / Wf, pack all weights into bf16 WMMA B-tile layout (L2-resident).
  cast_f32_bf16<<<1024, 256, 0, stream>>>(x, Xb, (long long)SEQ * IN_DIM);
  cast_f32_bf16<<<1024, 256, 0, stream>>>(Wf, Wfb, (long long)WIDTH * (HID_DIM + OUT_DIM));
  pack_wmma_b<<<1024, 256, 0, stream>>>(W0, 0, 0, WIDTH, IN_DIM, WIDTH, pW0x);
  pack_wmma_b<<<1024, 256, 0, stream>>>(W0, IN_DIM, 0, WIDTH, HID_DIM, WIDTH, pW0h);
  for (int i = 0; i < DEPTH; ++i)
    pack_wmma_b<<<1024, 256, 0, stream>>>(Wmid + (size_t)i * WIDTH * WIDTH, 0, 0, WIDTH,
                                          WIDTH, WIDTH, pWm + (size_t)i * WIDTH * WIDTH);
  pack_wmma_b<<<1024, 256, 0, stream>>>(Wf, 0, HID_DIM, HID_DIM + OUT_DIM, WIDTH, OUT_DIM, pWfo);

  // C = X @ W0[:1024,:] + b0   [8192 x 2048]  (batched WMMA GEMM)
  wmma_gemm_bf16<<<(SEQ / 16) * (WIDTH / 16) / 8, 256, 0, stream>>>(
      Xb, IN_DIM, pW0x, b0, Cb, WIDTH, WIDTH / 16, IN_DIM / 32);

  // Wfused = Wf[:, :1024] @ W0[1024:, :]   [2048 x 2048]  (WMMA GEMM, K=1024)
  wmma_gemm_bf16<<<(WIDTH / 16) * (WIDTH / 16) / 8, 256, 0, stream>>>(
      Wfb, HID_DIM + OUT_DIM, pW0h, nullptr, Wfus, WIDTH, WIDTH / 16, HID_DIM / 32);
  pack_wmma_b<<<1024, 256, 0, stream>>>(Wfus, 0, 0, WIDTH, WIDTH, WIDTH, pWfus);

  // d = bf[:1024] @ W0[1024:, :], folded into C rows t>=1.
  bias_w0h<<<WIDTH / 256, 256, 0, stream>>>(W0, bf, dvec);
  add_d<<<2048, 256, 0, stream>>>(Cb, dvec);

  // Sequential recurrence: 8192 steps x 4 uniform fused layers, 16 persistent blocks.
  rnn_sequential<<<NWG, 256, 0, stream>>>(Cb, pWfus, pWm, bmid, Zb, S0, S1, Rb,
                                          bar_cnt, bar_gen);

  // hidden_final = R[8191] @ Wf[:, :1024] + bf[:1024]  -> d_out[0:1024]
  final_hidden<<<HID_DIM / 256, 256, 0, stream>>>(Rb + (size_t)(SEQ - 1) * WIDTH, Wf, bf, out);

  // outputs = R @ Wf[:, 1024:] + bf[1024:]  [8192 x 1024]  (batched WMMA GEMM)
  wmma_gemm_bf16<<<(SEQ / 16) * (OUT_DIM / 16) / 8, 256, 0, stream>>>(
      Rb, WIDTH, pWfo, bf + HID_DIM, out + HID_DIM, OUT_DIM, OUT_DIM / 16, WIDTH / 32);
}
